// SE3TransformerTrIP_20564303413649
// MI455X (gfx1250) — matrix operand order repaired and checked by reading, not compiled, not run on hardware
//
#include <hip/hip_runtime.h>
#include <hip/hip_bf16.h>
#include <math.h>

typedef _Float16 half_t;
typedef __attribute__((ext_vector_type(16))) _Float16 v16h;
typedef __attribute__((ext_vector_type(8)))  float    v8f;

#define N_NODES 8192
#define N_EDGES 32768
#define C_CH    32
#define KV_     8
#define KVT_    16
#define H_HEADS 8
#define OUT_CH  16
#define MID_    64
#define RAD_OUT_ 3072   /* 6 * KVT * C */
#define FIN_OUT_ 1024   /* 2 * OUT * C */
#define L_LAYERS 2
#define EPB     64      /* edges per block (4 WMMA M-tiles) */

union AFrag { v16h v; half_t h[16]; };
union CFrag { v8f  v; float  f[8];  };

// ---------------------------------------------------------------------------
// Edge prep: rn, rhat, es = [rn, edge_feats_0]
// ---------------------------------------------------------------------------
__global__ void prep_edges(const float* __restrict__ relp,
                           const float* __restrict__ ef0,
                           float* __restrict__ es, float* __restrict__ rhat) {
  int e = blockIdx.x * blockDim.x + threadIdx.x;
  if (e >= N_EDGES) return;
  float x = relp[e*3+0], y = relp[e*3+1], z = relp[e*3+2];
  float rn = sqrtf(x*x + y*y + z*z);
  float inv = 1.0f / (rn + 1e-12f);
  rhat[e*3+0] = x*inv; rhat[e*3+1] = y*inv; rhat[e*3+2] = z*inv;
  es[e*5+0] = rn;
  es[e*5+1] = ef0[e*4+0]; es[e*5+2] = ef0[e*4+1];
  es[e*5+3] = ef0[e*4+2]; es[e*5+4] = ef0[e*4+3];
}

// ---------------------------------------------------------------------------
// Convert W2 (f32, [64, nout]) into f16 WMMA B-fragment order:
//   idx = ((tile*2 + kchunk)*32 + lane)*16 + i  ->  W2[K*nout + col]
//   K = kchunk*32 + (lane>>4)*16 + i, col = tile*16 + (lane&15)
// ---------------------------------------------------------------------------
__global__ void cvt_w2_frag(const float* __restrict__ W2, half_t* __restrict__ W2h, int nout) {
  int idx = blockIdx.x * blockDim.x + threadIdx.x;
  int total = 64 * nout;
  if (idx >= total) return;
  int i    = idx & 15;
  int lane = (idx >> 4) & 31;
  int ck   = idx >> 9;
  int t    = ck >> 1, kc = ck & 1;
  int n    = lane & 15, hs = lane >> 4;
  int K    = kc*32 + hs*16 + i;
  int col  = t*16 + n;
  W2h[idx] = (half_t)W2[(size_t)K * nout + col];
}

__global__ void copy_f32(const float* __restrict__ s, float* __restrict__ d, int n) {
  int i = blockIdx.x * blockDim.x + threadIdx.x;
  if (i < n) d[i] = s[i];
}
__global__ void zero_f32(float* __restrict__ p, int n) {
  int i = blockIdx.x * blockDim.x + threadIdx.x;
  if (i < n) p[i] = 0.0f;
}

// ---------------------------------------------------------------------------
// Fused ConvSE3: 64-edge tile per block (one wave). Radial MLP (LN+ReLU) ->
// four 16x64 A fragments in VGPRs -> per B column tile: one fragment-ordered
// B load feeds 8 v_wmma_f32_16x16x32_f16 (4 M-tiles x 2 k-steps) -> D tiles
// parked in LDS -> full-width (32-lane, 2 edges/lane) CG contraction.
// R never hits HBM; B L2 traffic amortized 4x vs 16-edge tiles.
// FINAL=false: writes m0 [E,16], m1 [E,16,3] (scaled).
// FINAL=true : atomically scatters mf into out [N,16].
// ---------------------------------------------------------------------------
template <bool FINAL>
__global__ __launch_bounds__(32)
void conv_se3(const float* __restrict__ es, const float* __restrict__ rhat_g,
              const float* __restrict__ scale, const int* __restrict__ src,
              const int* __restrict__ dst,
              const float* __restrict__ W1, const float* __restrict__ bias1,
              const float* __restrict__ gam, const float* __restrict__ bet,
              const half_t* __restrict__ W2h,
              const float* __restrict__ f0, const float* __restrict__ f1,
              float* __restrict__ m0, float* __restrict__ m1, int nout) {
  __shared__ float  hF[EPB][64];
  __shared__ half_t hH[EPB][64];
  __shared__ float  f0s[EPB][32];
  __shared__ float  f1s[EPB][32][3];
  __shared__ float  Qe[EPB][3][3][3];
  __shared__ float  Rt[4][16][16];
  __shared__ float  acc0[EPB][16];
  __shared__ float  acc1[EPB][16][3];
  __shared__ float  rh[EPB][3];
  __shared__ float  sc[EPB];
  __shared__ float  mu_s[EPB], iv_s[EPB];

  const int lane = threadIdx.x;
  const int eb   = blockIdx.x * EPB;

  // Per-edge scalars + CG basis (2 edges per lane)
  for (int el = lane; el < EPB; el += 32) {
    int e = eb + el;
    float r0 = rhat_g[e*3+0], r1 = rhat_g[e*3+1], r2 = rhat_g[e*3+2];
    rh[el][0] = r0; rh[el][1] = r1; rh[el][2] = r2;
    sc[el] = scale[e];
    if (!FINAL) {
      const float is3  = 0.57735026919f;   // 1/sqrt(3)
      const float is2  = 0.70710678118f;   // 1/sqrt(2)
      const float sq15 = 1.22474487139f;   // sqrt(1.5)
      float r[3] = {r0, r1, r2};
      for (int a = 0; a < 3; a++)
        for (int b = 0; b < 3; b++) {
          Qe[el][0][a][b] = (a == b) ? is3 : 0.0f;
          Qe[el][2][a][b] = sq15 * (r[a]*r[b] - ((a == b) ? (1.0f/3.0f) : 0.0f));
        }
      Qe[el][1][0][0] = 0.0f;    Qe[el][1][0][1] =  r2*is2; Qe[el][1][0][2] = -r1*is2;
      Qe[el][1][1][0] = -r2*is2; Qe[el][1][1][1] = 0.0f;    Qe[el][1][1][2] =  r0*is2;
      Qe[el][1][2][0] =  r1*is2; Qe[el][1][2][1] = -r0*is2; Qe[el][1][2][2] = 0.0f;
    }
  }
  // Gather source-node features
  for (int idx = lane; idx < EPB*32; idx += 32) {
    int el = idx >> 5, c = idx & 31;
    int s  = src[eb + el];
    f0s[el][c]    = f0[(size_t)s*32 + c];
    f1s[el][c][0] = f1[(size_t)s*96 + c*3 + 0];
    f1s[el][c][1] = f1[(size_t)s*96 + c*3 + 1];
    f1s[el][c][2] = f1[(size_t)s*96 + c*3 + 2];
  }
  // Radial layer 1: h = es @ W1 + b1    ([EPB,5] x [5,64])
  for (int idx = lane; idx < EPB*64; idx += 32) {
    int el = idx >> 6, j = idx & 63;
    float a = bias1[j];
    for (int k = 0; k < 5; k++) a += es[(size_t)(eb + el)*5 + k] * W1[k*64 + j];
    hF[el][j] = a;
  }
  __syncthreads();
  // LayerNorm stats per edge row
  for (int el = lane; el < EPB; el += 32) {
    float mu = 0.0f;
    for (int j = 0; j < 64; j++) mu += hF[el][j];
    mu *= (1.0f/64.0f);
    float v = 0.0f;
    for (int j = 0; j < 64; j++) { float d = hF[el][j] - mu; v += d*d; }
    v *= (1.0f/64.0f);
    mu_s[el] = mu; iv_s[el] = rsqrtf(v + 1e-5f);
  }
  __syncthreads();
  // LN * g + b, ReLU, cast f16
  for (int idx = lane; idx < EPB*64; idx += 32) {
    int el = idx >> 6, j = idx & 63;
    float v = (hF[el][j] - mu_s[el]) * iv_s[el] * gam[j] + bet[j];
    hH[el][j] = (half_t)fmaxf(v, 0.0f);
  }
  // Zero accumulators
  for (int idx = lane; idx < EPB*16; idx += 32) acc0[idx >> 4][idx & 15] = 0.0f;
  if (!FINAL)
    for (int idx = lane; idx < EPB*48; idx += 32) {
      int el = idx / 48, r = idx % 48;
      acc1[el][r/3][r%3] = 0.0f;
    }
  __syncthreads();

  // Build 4 A fragments (16x64 f16 each, as 2 k-chunks of 16x32): lane holds row m=lane&15 of each
  // 16-edge sub-tile; half selects K bank.
  const int mrow = lane & 15, hs = lane >> 4;
  AFrag af[4][2];
  for (int s = 0; s < 4; s++)
    for (int kc = 0; kc < 2; kc++)
      for (int i = 0; i < 16; i++) {
        int K = kc*32 + hs*16 + i;
        af[s][kc].h[i] = hH[s*16 + mrow][K];
      }

  const int ntiles = nout >> 4;
  for (int t = 0; t < ntiles; t++) {
    const half_t* bp = W2h + (size_t)(t*2) * 512;
    v16h b0 = *(const v16h*)(bp + (size_t)lane * 16);
    v16h b1 = *(const v16h*)(bp + 512 + (size_t)lane * 16);
    if (t + 1 < ntiles) __builtin_prefetch(bp + 1024 + (size_t)lane * 16, 0, 1);

    // 8 WMMAs per B load: 4 M-tiles x 2 k-steps
    for (int s = 0; s < 4; s++) {
      CFrag acc; acc.v = (v8f){};
      acc.v = __builtin_amdgcn_wmma_f32_16x16x32_f16(false, af[s][0].v, false, b0, (short)0, acc.v, false, false);
      acc.v = __builtin_amdgcn_wmma_f32_16x16x32_f16(false, af[s][1].v, false, b1, (short)0, acc.v, false, false);
      // D layout: lanes 0-15 -> M=g, lanes 16-31 -> M=g+8; N = lane&15
      for (int g = 0; g < 8; g++) Rt[s][hs*8 + g][lane & 15] = acc.f[g];
    }
    __syncthreads();

    // Contract this 16-col tile of R against gathered features. 16 aligned
    // columns share one output index o and half of the C=32 channel range.
    // All 32 lanes active: 2 edges per lane.
    {
      int n0 = t << 4;
      for (int el = lane; el < EPB; el += 32) {
        int s4 = el >> 4, er = el & 15;
        const float* rrow = &Rt[s4][er][0];
        if (!FINAL) {
          if (n0 < 512) {                         // R00: m0 += R * f0s
            int o = n0 >> 5, cb = n0 & 31;
            float s = 0.0f;
            for (int n = 0; n < 16; n++) s += rrow[n] * f0s[el][cb + n];
            acc0[el][o] += s;
          } else if (n0 < 1024) {                 // R01: m1 += R * f0s * rhat
            int j0 = n0 - 512, o = j0 >> 5, cb = j0 & 31;
            float s = 0.0f;
            for (int n = 0; n < 16; n++) s += rrow[n] * f0s[el][cb + n];
            acc1[el][o][0] += rh[el][0]*s; acc1[el][o][1] += rh[el][1]*s; acc1[el][o][2] += rh[el][2]*s;
          } else if (n0 < 1536) {                 // R10: m0 += R * (rhat . f1s)
            int j0 = n0 - 1024, o = j0 >> 5, cb = j0 & 31;
            float s = 0.0f;
            for (int n = 0; n < 16; n++) {
              int c = cb + n;
              float d = rh[el][0]*f1s[el][c][0] + rh[el][1]*f1s[el][c][1] + rh[el][2]*f1s[el][c][2];
              s += rrow[n] * d;
            }
            acc0[el][o] += s;
          } else {                                // R11: m1 += R * (Q[jdx] @ f1s)
            int j0 = n0 - 1536;
            int jdx = j0 >> 9, rem = j0 & 511, o = rem >> 5, cb = rem & 31;
            float s0 = 0.0f, s1 = 0.0f, s2 = 0.0f;
            for (int n = 0; n < 16; n++) {
              int c = cb + n;
              float fx = f1s[el][c][0], fy = f1s[el][c][1], fz = f1s[el][c][2];
              float r  = rrow[n];
              s0 += r * (Qe[el][jdx][0][0]*fx + Qe[el][jdx][0][1]*fy + Qe[el][jdx][0][2]*fz);
              s1 += r * (Qe[el][jdx][1][0]*fx + Qe[el][jdx][1][1]*fy + Qe[el][jdx][1][2]*fz);
              s2 += r * (Qe[el][jdx][2][0]*fx + Qe[el][jdx][2][1]*fy + Qe[el][jdx][2][2]*fz);
            }
            acc1[el][o][0] += s0; acc1[el][o][1] += s1; acc1[el][o][2] += s2;
          }
        } else {
          if (n0 < 512) {                         // Rf00
            int o = n0 >> 5, cb = n0 & 31;
            float s = 0.0f;
            for (int n = 0; n < 16; n++) s += rrow[n] * f0s[el][cb + n];
            acc0[el][o] += s;
          } else {                                // Rf10
            int j0 = n0 - 512, o = j0 >> 5, cb = j0 & 31;
            float s = 0.0f;
            for (int n = 0; n < 16; n++) {
              int c = cb + n;
              float d = rh[el][0]*f1s[el][c][0] + rh[el][1]*f1s[el][c][1] + rh[el][2]*f1s[el][c][2];
              s += rrow[n] * d;
            }
            acc0[el][o] += s;
          }
        }
      }
    }
    __syncthreads();
  }

  // Write back (scaled)
  if (!FINAL) {
    for (int idx = lane; idx < EPB*16; idx += 32) {
      int el = idx >> 4, o = idx & 15;
      m0[(size_t)(eb + el)*16 + o] = acc0[el][o] * sc[el];
    }
    for (int idx = lane; idx < EPB*48; idx += 32) {
      int el = idx / 48, r = idx % 48;
      m1[(size_t)(eb + el)*48 + r] = acc1[el][r/3][r%3] * sc[el];
    }
  } else {
    for (int idx = lane; idx < EPB*16; idx += 32) {
      int el = idx >> 4, o = idx & 15;
      atomicAdd(&m0[(size_t)dst[eb + el]*16 + o], acc0[el][o] * sc[el]);
    }
  }
}

// ---------------------------------------------------------------------------
// Queries, packed per head: qp[n*32 + h*4 + d]
// ---------------------------------------------------------------------------
__global__ void q_kernel(const float* __restrict__ f0, const float* __restrict__ f1,
                         const float* __restrict__ Wq0, const float* __restrict__ Wq1,
                         float* __restrict__ qp) {
  int idx = blockIdx.x * blockDim.x + threadIdx.x;
  if (idx >= N_NODES * KV_) return;
  int n = idx >> 3, k = idx & 7;
  float q0 = 0.0f, qx = 0.0f, qy = 0.0f, qz = 0.0f;
  for (int c = 0; c < 32; c++) {
    float w0 = Wq0[c*8 + k], w1 = Wq1[c*8 + k];
    q0 += f0[(size_t)n*32 + c] * w0;
    qx += f1[(size_t)n*96 + c*3 + 0] * w1;
    qy += f1[(size_t)n*96 + c*3 + 1] * w1;
    qz += f1[(size_t)n*96 + c*3 + 2] * w1;
  }
  qp[(size_t)n*32 + k]           = q0;
  qp[(size_t)n*32 + 8 + k*3 + 0] = qx;
  qp[(size_t)n*32 + 8 + k*3 + 1] = qy;
  qp[(size_t)n*32 + 8 + k*3 + 2] = qz;
}

__device__ inline unsigned fkey(float f) {
  unsigned u = __float_as_uint(f);
  return (u & 0x80000000u) ? ~u : (u | 0x80000000u);
}
__device__ inline float fdec(unsigned k) {
  return (k & 0x80000000u) ? __uint_as_float(k ^ 0x80000000u) : __uint_as_float(~k);
}

__global__ void attn_logit(const float* __restrict__ m0, const float* __restrict__ m1,
                           const float* __restrict__ qp, const int* __restrict__ dst,
                           float* __restrict__ logit, unsigned* __restrict__ segmax) {
  int idx = blockIdx.x * blockDim.x + threadIdx.x;
  if (idx >= N_EDGES * H_HEADS) return;
  int e = idx >> 3, h = idx & 7;
  int d = dst[e];
  float s = 0.0f;
  for (int dd = 0; dd < 4; dd++) {
    int i = h*4 + dd;
    float kv = (i < 8) ? m0[(size_t)e*16 + i] : m1[(size_t)e*48 + (i - 8)];
    s += kv * qp[(size_t)d*32 + i];
  }
  s *= 0.5f;                           // 1/sqrt(4)
  logit[idx] = s;
  atomicMax(&segmax[(size_t)d*8 + h], fkey(s));
}

__global__ void attn_exp(float* __restrict__ logit, const unsigned* __restrict__ segmax,
                         const int* __restrict__ dst, float* __restrict__ den) {
  int idx = blockIdx.x * blockDim.x + threadIdx.x;
  if (idx >= N_EDGES * H_HEADS) return;
  int e = idx >> 3, h = idx & 7;
  int d = dst[e];
  float z = __expf(logit[idx] - fdec(segmax[(size_t)d*8 + h]));
  logit[idx] = z;
  atomicAdd(&den[(size_t)d*8 + h], z);
}

__global__ void attn_out(const float* __restrict__ z, const float* __restrict__ den,
                         const int* __restrict__ dst, const float* __restrict__ m0,
                         const float* __restrict__ m1, float* __restrict__ o0,
                         float* __restrict__ o1) {
  int idx = blockIdx.x * blockDim.x + threadIdx.x;
  if (idx >= N_EDGES * H_HEADS) return;
  int e = idx >> 3, h = idx & 7;
  int d = dst[e];
  float a = z[idx] / (den[(size_t)d*8 + h] + 1e-9f);
  atomicAdd(&o0[(size_t)d*8 + h], a * m0[(size_t)e*16 + 8 + h]);
  for (int ax = 0; ax < 3; ax++)
    atomicAdd(&o1[(size_t)d*24 + h*3 + ax], a * m1[(size_t)e*48 + (8 + h)*3 + ax]);
}

// ---------------------------------------------------------------------------
// Projection (concat attn-out with skip) + NormSE3 gate. One node per thread.
// ---------------------------------------------------------------------------
__global__ __launch_bounds__(64)
void proj_norm(const float* __restrict__ o0, const float* __restrict__ o1,
               const float* __restrict__ f0, const float* __restrict__ f1,
               const float* __restrict__ Wp0, const float* __restrict__ Wp1,
               const float* __restrict__ ng0, const float* __restrict__ nb0,
               const float* __restrict__ ng1, const float* __restrict__ nb1,
               float* __restrict__ f0o, float* __restrict__ f1o) {
  __shared__ float buf[64][128];
  int t = threadIdx.x;
  int n = blockIdx.x * 64 + t;

  for (int o = 0; o < 32; o++) {
    float a0 = 0.0f;
    for (int c = 0; c < 8; c++)  a0 += o0[(size_t)n*8 + c] * Wp0[c*32 + o];
    for (int c = 0; c < 32; c++) a0 += f0[(size_t)n*32 + c] * Wp0[(8 + c)*32 + o];
    buf[t][o] = a0;
    float ax = 0.0f, ay = 0.0f, az = 0.0f;
    for (int c = 0; c < 8; c++) {
      float w = Wp1[c*32 + o];
      ax += o1[(size_t)n*24 + c*3 + 0] * w;
      ay += o1[(size_t)n*24 + c*3 + 1] * w;
      az += o1[(size_t)n*24 + c*3 + 2] * w;
    }
    for (int c = 0; c < 32; c++) {
      float w = Wp1[(8 + c)*32 + o];
      ax += f1[(size_t)n*96 + c*3 + 0] * w;
      ay += f1[(size_t)n*96 + c*3 + 1] * w;
      az += f1[(size_t)n*96 + c*3 + 2] * w;
    }
    buf[t][32 + o*3 + 0] = ax; buf[t][32 + o*3 + 1] = ay; buf[t][32 + o*3 + 2] = az;
  }

  float nrm[32];
  // degree-0 gate
  float mu = 0.0f;
  for (int o = 0; o < 32; o++) { nrm[o] = sqrtf(buf[t][o]*buf[t][o] + 1e-12f); mu += nrm[o]; }
  mu *= (1.0f/32.0f);
  float var = 0.0f;
  for (int o = 0; o < 32; o++) { float d = nrm[o] - mu; var += d*d; }
  var *= (1.0f/32.0f);
  float inv = rsqrtf(var + 1e-5f);
  for (int o = 0; o < 32; o++) {
    float g = fmaxf((nrm[o] - mu)*inv*ng0[o] + nb0[o], 0.0f) / nrm[o];
    f0o[(size_t)n*32 + o] = buf[t][o] * g;
  }
  // degree-1 gate
  mu = 0.0f;
  for (int o = 0; o < 32; o++) {
    float x = buf[t][32 + o*3 + 0], y = buf[t][32 + o*3 + 1], z = buf[t][32 + o*3 + 2];
    nrm[o] = sqrtf(x*x + y*y + z*z + 1e-12f); mu += nrm[o];
  }
  mu *= (1.0f/32.0f);
  var = 0.0f;
  for (int o = 0; o < 32; o++) { float d = nrm[o] - mu; var += d*d; }
  var *= (1.0f/32.0f);
  inv = rsqrtf(var + 1e-5f);
  for (int o = 0; o < 32; o++) {
    float g = fmaxf((nrm[o] - mu)*inv*ng1[o] + nb1[o], 0.0f) / nrm[o];
    f1o[(size_t)n*96 + o*3 + 0] = buf[t][32 + o*3 + 0] * g;
    f1o[(size_t)n*96 + o*3 + 1] = buf[t][32 + o*3 + 1] * g;
    f1o[(size_t)n*96 + o*3 + 2] = buf[t][32 + o*3 + 2] * g;
  }
}

// ---------------------------------------------------------------------------
extern "C" void kernel_launch(void* const* d_in, const int* in_sizes, int n_in,
                              void* d_out, int out_size, void* d_ws, size_t ws_size,
                              hipStream_t stream) {
  const float* node0 = (const float*)d_in[0];
  const float* node1 = (const float*)d_in[1];
  const float* ef0   = (const float*)d_in[2];
  const float* relp  = (const float*)d_in[3];
  const float* scale = (const float*)d_in[4];
  const int*   src   = (const int*)d_in[5];
  const int*   dst   = (const int*)d_in[6];
  const float* radW1 = (const float*)d_in[7];
  const float* radb1 = (const float*)d_in[8];
  const float* radg  = (const float*)d_in[9];
  const float* radbe = (const float*)d_in[10];
  const float* radW2 = (const float*)d_in[11];
  const float* Wq0   = (const float*)d_in[12];
  const float* Wq1   = (const float*)d_in[13];
  const float* Wp0   = (const float*)d_in[14];
  const float* Wp1   = (const float*)d_in[15];
  const float* ng0   = (const float*)d_in[16];
  const float* nb0   = (const float*)d_in[17];
  const float* ng1   = (const float*)d_in[18];
  const float* nb1   = (const float*)d_in[19];
  const float* finW1 = (const float*)d_in[20];
  const float* finb1 = (const float*)d_in[21];
  const float* fing  = (const float*)d_in[22];
  const float* finbe = (const float*)d_in[23];
  const float* finW2 = (const float*)d_in[24];

  char* base = (char*)d_ws;
  size_t off = 0;
  auto alloc = [&](size_t bytes) -> void* {
    void* p = base + off;
    off = (off + bytes + 255) & ~(size_t)255;
    return p;
  };
  float*  es_b   = (float*)alloc((size_t)N_EDGES*5*4);
  float*  rhat_b = (float*)alloc((size_t)N_EDGES*3*4);
  float*  f0a    = (float*)alloc((size_t)N_NODES*32*4);
  float*  f1a    = (float*)alloc((size_t)N_NODES*96*4);
  float*  f0b    = (float*)alloc((size_t)N_NODES*32*4);
  float*  f1b    = (float*)alloc((size_t)N_NODES*96*4);
  float*  m0_b   = (float*)alloc((size_t)N_EDGES*16*4);
  float*  m1_b   = (float*)alloc((size_t)N_EDGES*48*4);
  float*  qp_b   = (float*)alloc((size_t)N_NODES*32*4);
  float*  lg_b   = (float*)alloc((size_t)N_EDGES*8*4);
  unsigned* smax = (unsigned*)alloc((size_t)N_NODES*8*4);
  float*  den_b  = (float*)alloc((size_t)N_NODES*8*4);
  float*  o0_b   = (float*)alloc((size_t)N_NODES*8*4);
  float*  o1_b   = (float*)alloc((size_t)N_NODES*24*4);
  half_t* w2h    = (half_t*)alloc((size_t)L_LAYERS*64*RAD_OUT_*2);
  half_t* wfh    = (half_t*)alloc((size_t)64*FIN_OUT_*2);

  float* out = (float*)d_out;

  // ---- prep + weight conversion + feature init ----
  prep_edges<<<(N_EDGES + 255)/256, 256, 0, stream>>>(relp, ef0, es_b, rhat_b);
  for (int l = 0; l < L_LAYERS; l++)
    cvt_w2_frag<<<(64*RAD_OUT_ + 255)/256, 256, 0, stream>>>(
        radW2 + (size_t)l*64*RAD_OUT_, w2h + (size_t)l*64*RAD_OUT_, RAD_OUT_);
  cvt_w2_frag<<<(64*FIN_OUT_ + 255)/256, 256, 0, stream>>>(finW2, wfh, FIN_OUT_);
  copy_f32<<<(N_NODES*32 + 255)/256, 256, 0, stream>>>(node0, f0a, N_NODES*32);
  copy_f32<<<(N_NODES*96 + 255)/256, 256, 0, stream>>>(node1, f1a, N_NODES*96);
  zero_f32<<<(N_NODES*16 + 255)/256, 256, 0, stream>>>(out, N_NODES*16);

  float* f0c = f0a; float* f1c = f1a;
  float* f0n = f0b; float* f1n = f1b;

  for (int l = 0; l < L_LAYERS; l++) {
    conv_se3<false><<<N_EDGES/EPB, 32, 0, stream>>>(
        es_b, rhat_b, scale, src, dst,
        radW1 + (size_t)l*5*64, radb1 + (size_t)l*64, radg + (size_t)l*64, radbe + (size_t)l*64,
        w2h + (size_t)l*64*RAD_OUT_, f0c, f1c, m0_b, m1_b, RAD_OUT_);

    q_kernel<<<(N_NODES*KV_ + 255)/256, 256, 0, stream>>>(
        f0c, f1c, Wq0 + (size_t)l*32*8, Wq1 + (size_t)l*32*8, qp_b);

    zero_f32<<<(N_NODES*8 + 255)/256, 256, 0, stream>>>((float*)smax, N_NODES*8);
    zero_f32<<<(N_NODES*8 + 255)/256, 256, 0, stream>>>(den_b, N_NODES*8);
    zero_f32<<<(N_NODES*8 + 255)/256, 256, 0, stream>>>(o0_b, N_NODES*8);
    zero_f32<<<(N_NODES*24 + 255)/256, 256, 0, stream>>>(o1_b, N_NODES*24);

    attn_logit<<<(N_EDGES*8 + 255)/256, 256, 0, stream>>>(m0_b, m1_b, qp_b, dst, lg_b, smax);
    attn_exp<<<(N_EDGES*8 + 255)/256, 256, 0, stream>>>(lg_b, smax, dst, den_b);
    attn_out<<<(N_EDGES*8 + 255)/256, 256, 0, stream>>>(lg_b, den_b, dst, m0_b, m1_b, o0_b, o1_b);

    proj_norm<<<N_NODES/64, 64, 0, stream>>>(
        o0_b, o1_b, f0c, f1c,
        Wp0 + (size_t)l*40*32, Wp1 + (size_t)l*40*32,
        ng0 + (size_t)l*32, nb0 + (size_t)l*32, ng1 + (size_t)l*32, nb1 + (size_t)l*32,
        f0n, f1n);

    float* tf = f0c; f0c = f0n; f0n = tf;
    tf = f1c; f1c = f1n; f1n = tf;
  }

  // ---- final ConvSE3 -> degree-0 output, scattered by dst ----
  conv_se3<true><<<N_EDGES/EPB, 32, 0, stream>>>(
      es_b, rhat_b, scale, src, dst,
      finW1, finb1, fing, finbe, wfh, f0c, f1c, out, m1_b, FIN_OUT_);

  (void)in_sizes; (void)n_in; (void)out_size; (void)ws_size;
}
